// HypercorrSqueezeNetwork_imr_12111807775276
// MI455X (gfx1250) — matrix operand order repaired and checked by reading, not compiled, not run on hardware
//
#include <hip/hip_runtime.h>
#include <cstddef>

typedef __attribute__((ext_vector_type(16))) __bf16 v16bf;
typedef __attribute__((ext_vector_type(4)))  __bf16 v4bf;
typedef __attribute__((ext_vector_type(8)))  float  v8f;

#define EPSN 1e-12f
#define LDA 136   // 128 + 8 pad (bf16 units) -> 272B row stride, 16B aligned
#define LDH 72    // 64 + 8 pad

// ---------------- block reduction helper ----------------
__device__ __forceinline__ float blockReduceSum(float v, float* buf) {
  int tid = threadIdx.x;
  buf[tid] = v;
  __syncthreads();
  for (int s = 128; s > 0; s >>= 1) {
    if (tid < s) buf[tid] += buf[tid + s];
    __syncthreads();
  }
  float r = buf[0];
  __syncthreads();
  return r;
}

// ---------------- 1: mask + foreground count ----------------
__global__ void k_mask(const float* __restrict__ s_y,
                       float* __restrict__ mask, float* __restrict__ numf) {
  __shared__ float buf[256];
  int b = blockIdx.x;
  float cnt = 0.f;
  for (int n = threadIdx.x; n < 2500; n += 256) {
    int rr = n / 50, cc = n % 50;                       // iy = 8*rr, ix = 8*cc
    float v = s_y[b * 160000 + rr * 3200 + cc * 8];
    float m = (v == 1.0f) ? 1.f : 0.f;
    mask[b * 2500 + n] = m;
    cnt += m;
  }
  float t = blockReduceSum(cnt, buf);
  if (threadIdx.x == 0) numf[b] = t;
}

// ---------------- 2: semantic prototype ----------------
__global__ void k_sproto(const float* __restrict__ s_f, const float* __restrict__ mask,
                         const float* __restrict__ numf, float* __restrict__ sproto) {
  int b = blockIdx.x >> 2;
  int d = (blockIdx.x & 3) * 256 + threadIdx.x;
  const float* sf = s_f + (size_t)b * 2560000 + d;
  const float* mk = mask + b * 2500;
  float acc = 0.f;
  for (int n = 0; n < 2500; ++n) acc += mk[n] * sf[(size_t)n * 1024];
  sproto[b * 1024 + d] = acc / (numf[b] + 1e-4f);
}

// ---------------- 3: select + normalize prototype ----------------
__global__ void k_select(const float* __restrict__ proto, const float* __restrict__ sproto,
                         const float* __restrict__ numf,
                         float* __restrict__ P, float* __restrict__ sgn) {
  __shared__ float buf[256];
  __shared__ float simsh[16];
  __shared__ float ncol[16];
  __shared__ int sidx;
  int b = blockIdx.x, tid = threadIdx.x;
  const float* sp = sproto + b * 1024;

  float a = 0.f;
  for (int d = tid; d < 1024; d += 256) { float v = sp[d]; a += v * v; }
  float nsp = sqrtf(blockReduceSum(a, buf));

  for (int c = 0; c < 16; ++c) {
    float ad = 0.f, an = 0.f;
    for (int d = tid; d < 1024; d += 256) {
      float w = proto[d * 16 + c];
      ad += sp[d] * w;
      an += w * w;
    }
    float sd = blockReduceSum(ad, buf);
    float sn = blockReduceSum(an, buf);
    if (tid == 0) {
      float nc = sqrtf(sn);
      ncol[c] = nc;
      simsh[c] = sd / (fmaxf(nsp, EPSN) * fmaxf(nc, EPSN));
    }
    __syncthreads();
  }
  if (tid == 0) {
    int best = 0; float bv = simsh[0];
    for (int c = 1; c < 16; ++c) if (simsh[c] > bv) { bv = simsh[c]; best = c; }
    sidx = best;
    sgn[b] = (numf[b] > 0.5f) ? 1.f : numf[b];
  }
  __syncthreads();
  int idx = sidx;
  float nc = fmaxf(ncol[idx], EPSN);
  float at = 0.f;
  for (int d = tid; d < 1024; d += 256) { float t = proto[d * 16 + idx] / nc; at += t * t; }
  float nt = fmaxf(sqrtf(blockReduceSum(at, buf)), EPSN);
  for (int d = tid; d < 1024; d += 256)
    P[b * 1024 + d] = (proto[d * 16 + idx] / nc) / nt;
}

// ---------------- 4: weight conversion to bf16 ----------------
__global__ void k_cvt(const float* __restrict__ Wd_f, const float* __restrict__ Wu_f,
                      __bf16* __restrict__ Wd, __bf16* __restrict__ Wu) {
  int i = blockIdx.x * 256 + threadIdx.x;   // grid covers exactly 65536
  Wd[i] = (__bf16)Wd_f[i];
  Wu[i] = (__bf16)Wu_f[i];
}

// ---------------- 5: per-row enhance scale (wave per row) ----------------
__global__ void k_rowscale(const float* __restrict__ x, const float* __restrict__ s_f,
                           const float* __restrict__ P, const float* __restrict__ sgn,
                           float* __restrict__ sc) {
  int wid = threadIdx.x >> 5, lane = threadIdx.x & 31;
  int r = blockIdx.x * 8 + wid;
  const float* f; int b;
  if (r < 20000) { f = s_f + (size_t)r * 1024; b = r / 2500; }
  else           { f = x + (size_t)(r - 20000) * 1024; b = (r - 20000) / 2500; }
  const float* p = P + b * 1024;
  float n2 = 0.f, dt = 0.f;
#pragma unroll
  for (int i = 0; i < 8; ++i) {
    float4 v = *(const float4*)(f + i * 128 + lane * 4);
    float4 q = *(const float4*)(p + i * 128 + lane * 4);
    n2 += v.x * v.x + v.y * v.y + v.z * v.z + v.w * v.w;
    dt += v.x * q.x + v.y * q.y + v.z * q.z + v.w * q.w;
  }
#pragma unroll
  for (int m = 16; m; m >>= 1) {
    n2 += __shfl_xor(n2, m, 32);
    dt += __shfl_xor(dt, m, 32);
  }
  if (lane == 0) {
    float g = 32.0f * sgn[b] * dt / fmaxf(sqrtf(n2), EPSN);   // scale = sqrt(1024) = 32
    g = fminf(fmaxf(g, 0.f), 6.f);                            // ReLU6
    sc[r] = 1.f + g;
  }
}

// ---------------- 6: fused enhance*feat -> GEMM1 -> ReLU -> GEMM2 ----------------
__global__ __launch_bounds__(256) void k_mlp(
    const float* __restrict__ x, const float* __restrict__ s_f,
    const float* __restrict__ sc,
    const __bf16* __restrict__ Wd, const __bf16* __restrict__ Wu,
    const float* __restrict__ b_down, const float* __restrict__ b_up,
    float* __restrict__ out) {
  __shared__ __bf16 Alds[64 * LDA];
  __shared__ __bf16 Hlds[64 * LDH];
  int tid = threadIdx.x;
  int wid = tid >> 5, lane = tid & 31;
  int hh = lane >> 4, lm = lane & 15;
  int R0 = blockIdx.x * 64;

  // GEMM1: 64x64 tile = 16 (4x4) 16x16 WMMA tiles; wave owns tiles wid and wid+8
  int mi0 = wid >> 2, ni0 = wid & 3;
  int mi1 = (wid + 8) >> 2, ni1 = (wid + 8) & 3;
  v8f c0 = {0.f, 0.f, 0.f, 0.f, 0.f, 0.f, 0.f, 0.f};
  v8f c1 = c0;

  for (int kc = 0; kc < 8; ++kc) {       // K = 1024 in chunks of 128
    int k0 = kc * 128;
    // cooperative load + row-scale + bf16 convert of A chunk [64 x 128]
#pragma unroll
    for (int it = 0; it < 8; ++it) {
      int idx = it * 1024 + tid * 4;
      int row = idx >> 7;
      int kk = idx & 127;
      int r = R0 + row;
      const float* f = (r < 20000) ? (s_f + (size_t)r * 1024)
                                   : (x + (size_t)(r - 20000) * 1024);
      float4 v = *(const float4*)(f + k0 + kk);
      float s = sc[r];
      v4bf bv;
      bv[0] = (__bf16)(v.x * s); bv[1] = (__bf16)(v.y * s);
      bv[2] = (__bf16)(v.z * s); bv[3] = (__bf16)(v.w * s);
      *(v4bf*)(&Alds[row * LDA + kk]) = bv;
    }
    __syncthreads();
#pragma unroll
    for (int s = 0; s < 4; ++s) {        // 4 k-steps of 32
      v16bf a0, a1;
      const __bf16* base0 = &Alds[(mi0 * 16 + lm) * LDA + s * 32];
      const __bf16* base1 = &Alds[(mi1 * 16 + lm) * LDA + s * 32];
#pragma unroll
      for (int i = 0; i < 8; ++i) {      // A layout: K = 8h+i and 16+8h+i
        a0[i]     = base0[8 * hh + i];
        a0[8 + i] = base0[16 + 8 * hh + i];
        a1[i]     = base1[8 * hh + i];
        a1[8 + i] = base1[16 + 8 * hh + i];
      }
      int kg = k0 + s * 32;
      v16bf b0, b1;                      // B layout: col = lm, K = 8h+i and 16+8h+i
      const __bf16* bp0 = Wd + (size_t)(ni0 * 16 + lm) * 1024 + kg + 8 * hh;
      const __bf16* bp1 = Wd + (size_t)(ni1 * 16 + lm) * 1024 + kg + 8 * hh;
#pragma unroll
      for (int i = 0; i < 8; ++i) { b0[i] = bp0[i]; b0[8 + i] = bp0[16 + i]; b1[i] = bp1[i]; b1[8 + i] = bp1[16 + i]; }
      c0 = __builtin_amdgcn_wmma_f32_16x16x32_bf16(false, a0, false, b0, (short)0, c0, false, false);
      c1 = __builtin_amdgcn_wmma_f32_16x16x32_bf16(false, a1, false, b1, (short)0, c1, false, false);
    }
    __syncthreads();
  }

  // bias + ReLU + stage hmid (bf16) to LDS; C layout: row = g + 8h, col = lm
  {
    float bd0 = b_down[ni0 * 16 + lm];
    float bd1 = b_down[ni1 * 16 + lm];
#pragma unroll
    for (int g = 0; g < 8; ++g) {
      int r0 = mi0 * 16 + g + 8 * hh;
      int r1 = mi1 * 16 + g + 8 * hh;
      Hlds[r0 * LDH + ni0 * 16 + lm] = (__bf16)fmaxf(c0[g] + bd0, 0.f);
      Hlds[r1 * LDH + ni1 * 16 + lm] = (__bf16)fmaxf(c1[g] + bd1, 0.f);
    }
  }
  __syncthreads();

  // GEMM2: [64 x 64] @ [64 x 1024]; wave handles n-tiles wid+8j, all 4 m-tiles
  for (int mi = 0; mi < 4; ++mi) {
    v16bf a0, a1;                        // k-steps K=0..31 and K=32..63
    const __bf16* base = &Hlds[(mi * 16 + lm) * LDH];
#pragma unroll
    for (int i = 0; i < 8; ++i) {
      a0[i]     = base[8 * hh + i];
      a0[8 + i] = base[16 + 8 * hh + i];
      a1[i]     = base[32 + 8 * hh + i];
      a1[8 + i] = base[48 + 8 * hh + i];
    }
    for (int j = 0; j < 8; ++j) {
      int nt = wid + 8 * j;
      int col = nt * 16 + lm;
      float bu = b_up[col];
      v8f c = {bu, bu, bu, bu, bu, bu, bu, bu};   // bias depends only on col
      const __bf16* bp0 = Wu + (size_t)col * 64 + 8 * hh;
      const __bf16* bp1 = bp0 + 32;
      v16bf b0, b1;
#pragma unroll
      for (int i = 0; i < 8; ++i) { b0[i] = bp0[i]; b0[8 + i] = bp0[16 + i]; b1[i] = bp1[i]; b1[8 + i] = bp1[16 + i]; }
      c = __builtin_amdgcn_wmma_f32_16x16x32_bf16(false, a0, false, b0, (short)0, c, false, false);
      c = __builtin_amdgcn_wmma_f32_16x16x32_bf16(false, a1, false, b1, (short)0, c, false, false);
      float* op = out + (size_t)(R0 + mi * 16 + 8 * hh) * 1024 + col;
#pragma unroll
      for (int g = 0; g < 8; ++g) op[(size_t)g * 1024] = c[g];
    }
  }
}

// ---------------- launch ----------------
extern "C" void kernel_launch(void* const* d_in, const int* in_sizes, int n_in,
                              void* d_out, int out_size, void* d_ws, size_t ws_size,
                              hipStream_t stream) {
  const float* x    = (const float*)d_in[0];   // [8,2500,1024]
  const float* s_f  = (const float*)d_in[1];   // [8,2500,1024]
  const float* s_y  = (const float*)d_in[2];   // [8,1,400,400]
  const float* prot = (const float*)d_in[3];   // [1024,16]
  const float* Wd_f = (const float*)d_in[4];   // [64,1024]
  const float* bd   = (const float*)d_in[5];   // [64]
  const float* Wu_f = (const float*)d_in[6];   // [1024,64]
  const float* bu   = (const float*)d_in[7];   // [1024]
  float* out = (float*)d_out;                  // [40000,1024]

  float* ws     = (float*)d_ws;
  float* mask   = ws;             // 20000
  float* numf   = mask + 20000;   // 8
  float* sproto = numf + 8;       // 8192
  float* P      = sproto + 8192;  // 8192 (16B aligned)
  float* sgn    = P + 8192;       // 8
  float* sc     = sgn + 8;        // 40000
  __bf16* Wd    = (__bf16*)(sc + 40000);  // 65536 bf16 (16B aligned)
  __bf16* Wu    = Wd + 65536;             // 65536 bf16

  k_mask    <<<8,    256, 0, stream>>>(s_y, mask, numf);
  k_sproto  <<<32,   256, 0, stream>>>(s_f, mask, numf, sproto);
  k_select  <<<8,    256, 0, stream>>>(prot, sproto, numf, P, sgn);
  k_cvt     <<<256,  256, 0, stream>>>(Wd_f, Wu_f, Wd, Wu);
  k_rowscale<<<5000, 256, 0, stream>>>(x, s_f, P, sgn, sc);
  k_mlp     <<<625,  256, 0, stream>>>(x, s_f, sc, Wd, Wu, bd, bu, out);
}
